// RegionProposalNetwork_5506148073793
// MI455X (gfx1250) — matrix operand order repaired and checked by reading, not compiled, not run on hardware
//
#include <hip/hip_runtime.h>
#include <hip/hip_bf16.h>
#include <math.h>

// ---------------------------------------------------------------------------
// RPN for MI455X (gfx1250): bf16 WMMA implicit-GEMM conv + heads, then
// device-side top-k / decode / NMS / compaction.
// ---------------------------------------------------------------------------

typedef __bf16 bf16_t;
typedef __attribute__((ext_vector_type(16))) __bf16 v16bf;
typedef __attribute__((ext_vector_type(8)))  float  v8f;

#define RPN_C 256
#define RPN_H 200
#define RPN_W 200
#define RPN_A 9
#define RPN_M (RPN_H * RPN_W)          // 40000 spatial positions
#define RPN_K (RPN_C * 9)              // 2304
#define RPN_FEATN (RPN_C * RPN_H * RPN_W)
#define RPN_NANCH (RPN_M * RPN_A)      // 360000
#define PRE_NMS 6000
#define POST_NMS 1000
#define NMS_T 0.7f
#define MIN_SIZE 1.0f
#define IMG_SZ 1600.0f
#define BBOX_CLIP 4.135166556742356f   // log(1000/16)
#define CAND_N 8192
#define MASK_W 188                     // ceil(6000/32)

union FragBF { v16bf v; uint4 q[2]; };

// ---------------------------------------------------------------------------
// Weight conversion kernels (f32 -> bf16)
// ---------------------------------------------------------------------------
__global__ void rpn_cvt_wconv(const float* __restrict__ w, bf16_t* __restrict__ wb, int n) {
    int i = blockIdx.x * blockDim.x + threadIdx.x;
    if (i < n) wb[i] = (bf16_t)w[i];
}

// combined 1x1 head weights: B2[n][c], n = 0..8 obj, 9..44 reg, 45..47 zero pad
__global__ void rpn_cvt_whead(const float* __restrict__ wobj, const float* __restrict__ wreg,
                              bf16_t* __restrict__ wb2) {
    int n = blockIdx.x;              // 0..47
    int c = threadIdx.x;             // 0..255
    float v = 0.0f;
    if (n < 9)       v = wobj[n * RPN_C + c];
    else if (n < 45) v = wreg[(n - 9) * RPN_C + c];
    wb2[n * RPN_C + c] = (bf16_t)v;
}

// k-table: for k = c*9+ky*3+kx, gather addr = baseOff + m, bounds from (dy,dx)
//   ktab[2k]   = c*H*W + (ky-1)*W + (kx-1)
//   ktab[2k+1] = ((ky-1) << 16) | ((kx-1) & 0xffff)
__global__ void rpn_build_ktab(int* __restrict__ ktab) {
    int k = blockIdx.x * blockDim.x + threadIdx.x;
    if (k >= RPN_K) return;
    int c  = k / 9;
    int t9 = k - c * 9;
    int ky = t9 / 3;
    int kx = t9 - ky * 3;
    int dy = ky - 1, dx = kx - 1;
    ktab[2 * k + 0] = c * RPN_H * RPN_W + dy * RPN_W + dx;
    ktab[2 * k + 1] = (dy << 16) | (dx & 0xffff);
}

// ---------------------------------------------------------------------------
// Kernel 1: 3x3 conv + bias + relu, implicit GEMM via v_wmma_f32_16x16x32_bf16
//   block = 32 spatial positions (2 M-subtiles), 8 waves x 2 N-tiles = 256 ch
//   -> 4 WMMAs per wave per K-chunk, B fragments reused across M-subtiles
//   staging is branchless: clamped address + post-load select, so all 4
//   feature gathers issue as a clause instead of exec-masked serial loads
// ---------------------------------------------------------------------------
__global__ void __launch_bounds__(256)
rpn_conv3x3(const float* __restrict__ feat, const bf16_t* __restrict__ wB,
            const int* __restrict__ ktab, const float* __restrict__ bconv,
            bf16_t* __restrict__ tB) {
    __shared__ __align__(16) bf16_t Alds[32 * 32];

    const int tid  = threadIdx.x;
    const int lane = tid & 31;
    const int wid  = tid >> 5;          // 0..7
    const int r    = lane & 15;
    const int half = lane >> 4;
    const int m0   = blockIdx.x * 32;
    const int n0   = wid * 32;          // two N-tiles: n0, n0+16

    // staging coords: fixed per thread (one row m, 4 consecutive k per chunk)
    const int ml   = tid >> 3;          // 0..31
    const int kgrp = (tid & 7) * 4;     // 0,4,...,28
    const int m    = m0 + ml;
    const int h    = m / RPN_W;
    const int w    = m - h * RPN_W;

    v8f c00 = {}, c01 = {}, c10 = {}, c11 = {};   // [msub][ntile]

    for (int kc = 0; kc < RPN_K / 32; ++kc) {
        // ---- cooperative stage of A chunk (32 M x 32 K) into LDS ----
        __syncthreads();
        {
            int k0 = kc * 32 + kgrp;
            const int4* tp = (const int4*)(ktab + (k0 << 1));
            int4 ta = tp[0];
            int4 tb = tp[1];
            int offs[4]  = { ta.x, ta.z, tb.x, tb.z };
            int packs[4] = { ta.y, ta.w, tb.y, tb.w };
            float vals[4];
            bool  oks[4];
            #pragma unroll
            for (int t = 0; t < 4; ++t) {
                int dy = packs[t] >> 16;
                int dx = (int)(short)(packs[t] & 0xffff);
                oks[t] = ((unsigned)(h + dy) < RPN_H) && ((unsigned)(w + dx) < RPN_W);
                int addr = offs[t] + m;
                addr = addr < 0 ? 0 : (addr > RPN_FEATN - 1 ? RPN_FEATN - 1 : addr);
                vals[t] = feat[addr];                 // unconditional clamped gather
            }
            union { bf16_t b[4]; uint2 u2; } pk;
            #pragma unroll
            for (int t = 0; t < 4; ++t)
                pk.b[t] = (bf16_t)(oks[t] ? vals[t] : 0.0f);
            *(uint2*)&Alds[ml * 32 + kgrp] = pk.u2;
        }
        __syncthreads();

        // ---- A fragments from LDS (b128 reads), two M-subtiles ----
        FragBF a0, a1;
        a0.q[0] = *(const uint4*)&Alds[r * 32 + half * 8];
        a0.q[1] = *(const uint4*)&Alds[r * 32 + 16 + half * 8];
        a1.q[0] = *(const uint4*)&Alds[(16 + r) * 32 + half * 8];
        a1.q[1] = *(const uint4*)&Alds[(16 + r) * 32 + 16 + half * 8];

        // ---- B fragments from global (two 8-K runs per lane: 8h, 16+8h) ----
        const bf16_t* wb0 = wB + (size_t)(n0 + r) * RPN_K + kc * 32 + half * 8;
        const bf16_t* wb1 = wB + (size_t)(n0 + 16 + r) * RPN_K + kc * 32 + half * 8;
        __builtin_prefetch(wb0 + 32, 0, 1);
        FragBF b0, b1;
        b0.q[0] = *(const uint4*)wb0;
        b0.q[1] = *(const uint4*)(wb0 + 16);
        b1.q[0] = *(const uint4*)wb1;
        b1.q[1] = *(const uint4*)(wb1 + 16);

        c00 = __builtin_amdgcn_wmma_f32_16x16x32_bf16(false, a0.v, false, b0.v,
                                                      (short)0, c00, false, false);
        c01 = __builtin_amdgcn_wmma_f32_16x16x32_bf16(false, a0.v, false, b1.v,
                                                      (short)0, c01, false, false);
        c10 = __builtin_amdgcn_wmma_f32_16x16x32_bf16(false, a1.v, false, b0.v,
                                                      (short)0, c10, false, false);
        c11 = __builtin_amdgcn_wmma_f32_16x16x32_bf16(false, a1.v, false, b1.v,
                                                      (short)0, c11, false, false);
    }

    // ---- epilogue: bias + relu + cvt bf16 store (t[m][n]) ----
    float ba = bconv[n0 + r];
    float bb = bconv[n0 + 16 + r];
    #pragma unroll
    for (int i = 0; i < 8; ++i) {
        int mA = m0 + i + half * 8;          // M-subtile 0
        int mB = m0 + 16 + i + half * 8;     // M-subtile 1
        tB[(size_t)mA * RPN_C + n0 + r]      = (bf16_t)fmaxf(c00[i] + ba, 0.0f);
        tB[(size_t)mA * RPN_C + n0 + 16 + r] = (bf16_t)fmaxf(c01[i] + bb, 0.0f);
        tB[(size_t)mB * RPN_C + n0 + r]      = (bf16_t)fmaxf(c10[i] + ba, 0.0f);
        tB[(size_t)mB * RPN_C + n0 + 16 + r] = (bf16_t)fmaxf(c11[i] + bb, 0.0f);
    }
}

// ---------------------------------------------------------------------------
// Kernel 2: 1x1 heads as GEMM: M=40000, K=256, N=48 (9 obj + 36 reg + pad)
//   block = 128 threads = 4 waves, wave w -> N-tile w
// ---------------------------------------------------------------------------
__global__ void __launch_bounds__(128)
rpn_heads(const bf16_t* __restrict__ tB, const bf16_t* __restrict__ wB2,
          const float* __restrict__ bobj, const float* __restrict__ breg,
          float* __restrict__ scores, float* __restrict__ deltas) {
    const int tid  = threadIdx.x;
    const int lane = tid & 31;
    const int wid  = tid >> 5;          // 0..3
    const int r    = lane & 15;
    const int half = lane >> 4;
    const int m0   = blockIdx.x * 16;
    const int n0   = wid * 16;

    v8f acc = {};
    #pragma unroll
    for (int kc = 0; kc < RPN_C / 32; ++kc) {
        FragBF a, b;
        const bf16_t* ta = tB + (size_t)(m0 + r) * RPN_C + kc * 32;
        a.q[0] = *(const uint4*)(ta + half * 8);
        a.q[1] = *(const uint4*)(ta + 16 + half * 8);
        const bf16_t* wb = wB2 + (size_t)(n0 + r) * RPN_C + kc * 32 + half * 8;
        b.q[0] = *(const uint4*)wb;
        b.q[1] = *(const uint4*)(wb + 16);
        acc = __builtin_amdgcn_wmma_f32_16x16x32_bf16(false, a.v, false, b.v,
                                                      (short)0, acc, false, false);
    }

    int n = n0 + r;
    #pragma unroll
    for (int i = 0; i < 8; ++i) {
        int mm = m0 + i + half * 8;
        float v = acc[i];
        if (n < 9) {
            scores[mm * RPN_A + n] = v + bobj[n];
        } else if (n < 45) {
            int t = n - 9;
            deltas[(mm * RPN_A + (t >> 2)) * 4 + (t & 3)] = v + breg[t];
        }
    }
}

// ---------------------------------------------------------------------------
// Top-k selection: orderable key, 256-bin histogram, threshold, compaction
// ---------------------------------------------------------------------------
__device__ __forceinline__ unsigned rpn_fkey(float f) {
    unsigned u = __float_as_uint(f);
    return (u & 0x80000000u) ? ~u : (u | 0x80000000u);
}

__global__ void rpn_sel_init(unsigned* hist, unsigned* cntrs, unsigned long long* cand) {
    int i = blockIdx.x * blockDim.x + threadIdx.x;
    if (i < 256) hist[i] = 0;
    if (i < 8)   cntrs[i] = 0;
    if (i < CAND_N) cand[i] = 0ull;
}

__global__ void rpn_hist(const float* __restrict__ scores, unsigned* hist) {
    int i = blockIdx.x * blockDim.x + threadIdx.x;
    if (i < RPN_NANCH) atomicAdd(&hist[rpn_fkey(scores[i]) >> 24], 1u);
}

__global__ void rpn_thresh(const unsigned* __restrict__ hist, unsigned* cntrs) {
    if (threadIdx.x == 0 && blockIdx.x == 0) {
        unsigned cnt = 0, bin = 0;
        for (int b = 255; b >= 0; --b) {
            if (cnt + hist[b] >= PRE_NMS) { bin = (unsigned)b; break; }
            cnt += hist[b];
            if (b == 0) bin = 0;
        }
        cntrs[1] = bin;
    }
}

__global__ void rpn_compact(const float* __restrict__ scores, unsigned* cntrs,
                            unsigned long long* __restrict__ cand) {
    int i = blockIdx.x * blockDim.x + threadIdx.x;
    if (i >= RPN_NANCH) return;
    unsigned k = rpn_fkey(scores[i]);
    if ((k >> 24) >= cntrs[1]) {
        unsigned pos = atomicAdd(&cntrs[0], 1u);
        if (pos < CAND_N)
            cand[pos] = ((unsigned long long)k << 32) | (unsigned long long)i;
    }
}

__global__ void rpn_bitonic(unsigned long long* __restrict__ d, int j, int k) {
    int i = blockIdx.x * blockDim.x + threadIdx.x;     // 0..8191
    int p = i ^ j;
    if (p > i) {
        unsigned long long a = d[i], b = d[p];
        bool descBlock = ((i & k) == 0);
        if (descBlock ? (a < b) : (a > b)) { d[i] = b; d[p] = a; }
    }
}

// ---------------------------------------------------------------------------
// Decode + clip + validity for the sorted top-6000
// ---------------------------------------------------------------------------
__global__ void rpn_decode(const unsigned long long* __restrict__ cand,
                           const float* __restrict__ deltas,
                           const float* __restrict__ anchors,
                           float* __restrict__ boxes6k, int* __restrict__ valid6k) {
    int i = blockIdx.x * blockDim.x + threadIdx.x;
    if (i >= PRE_NMS) return;
    unsigned long long e = cand[i];
    unsigned key = (unsigned)(e >> 32);
    int idx = (int)(e & 0xffffffffull);

    float ax1 = anchors[idx * 4 + 0], ay1 = anchors[idx * 4 + 1];
    float ax2 = anchors[idx * 4 + 2], ay2 = anchors[idx * 4 + 3];
    float wa = ax2 - ax1, ha = ay2 - ay1;
    float xa = ax1 + 0.5f * wa, ya = ay1 + 0.5f * ha;
    float dx = deltas[idx * 4 + 0], dy = deltas[idx * 4 + 1];
    float dw = fminf(deltas[idx * 4 + 2], BBOX_CLIP);
    float dh = fminf(deltas[idx * 4 + 3], BBOX_CLIP);
    float cx = dx * wa + xa, cy = dy * ha + ya;
    float w = __expf(dw) * wa, h = __expf(dh) * ha;
    float x1 = fminf(fmaxf(cx - 0.5f * w, 0.0f), IMG_SZ);
    float y1 = fminf(fmaxf(cy - 0.5f * h, 0.0f), IMG_SZ);
    float x2 = fminf(fmaxf(cx + 0.5f * w, 0.0f), IMG_SZ);
    float y2 = fminf(fmaxf(cy + 0.5f * h, 0.0f), IMG_SZ);

    boxes6k[i * 4 + 0] = x1; boxes6k[i * 4 + 1] = y1;
    boxes6k[i * 4 + 2] = x2; boxes6k[i * 4 + 3] = y2;
    valid6k[i] = (key != 0u && (x2 - x1) >= MIN_SIZE && (y2 - y1) >= MIN_SIZE) ? 1 : 0;
}

// ---------------------------------------------------------------------------
// NMS: suppression bitmask (row i, words of 32 j), then single-block greedy
// ---------------------------------------------------------------------------
__global__ void rpn_nms_mask(const float* __restrict__ boxes6k, unsigned* __restrict__ maskw) {
    int i = blockIdx.x;
    int w = threadIdx.x;
    if (w >= MASK_W) return;
    float x1 = boxes6k[i * 4 + 0], y1 = boxes6k[i * 4 + 1];
    float x2 = boxes6k[i * 4 + 2], y2 = boxes6k[i * 4 + 3];
    float a1 = (x2 - x1) * (y2 - y1);
    unsigned bits = 0;
    #pragma unroll 4
    for (int b = 0; b < 32; ++b) {
        int j = w * 32 + b;
        if (j < PRE_NMS && j > i) {
            float bx1 = boxes6k[j * 4 + 0], by1 = boxes6k[j * 4 + 1];
            float bx2 = boxes6k[j * 4 + 2], by2 = boxes6k[j * 4 + 3];
            float ix = fmaxf(0.0f, fminf(x2, bx2) - fmaxf(x1, bx1));
            float iy = fmaxf(0.0f, fminf(y2, by2) - fmaxf(y1, by1));
            float inter = ix * iy;
            float a2 = (bx2 - bx1) * (by2 - by1);
            float iou = inter / fmaxf(a1 + a2 - inter, 1e-6f);
            if (iou > NMS_T) bits |= (1u << b);
        }
    }
    maskw[i * MASK_W + w] = bits;
}

__global__ void __launch_bounds__(256)
rpn_nms_greedy(const unsigned* __restrict__ maskw, const int* __restrict__ valid6k,
               int* __restrict__ keep) {
    __shared__ unsigned rem[MASK_W];
    int tid = threadIdx.x;
    if (tid < MASK_W) {
        unsigned bits = 0;
        for (int b = 0; b < 32; ++b) {
            int j = tid * 32 + b;
            if (j >= PRE_NMS || !valid6k[j]) bits |= (1u << b);
        }
        rem[tid] = bits;
    }
    __syncthreads();
    for (int i = 0; i < PRE_NMS; ++i) {
        unsigned removed = (rem[i >> 5] >> (i & 31)) & 1u;
        __syncthreads();
        if (!removed && tid < MASK_W) rem[tid] |= maskw[i * MASK_W + tid];
        if (tid == 0) keep[i] = removed ? 0 : 1;
        __syncthreads();
    }
}

__global__ void __launch_bounds__(256)
rpn_finalize(const int* __restrict__ keep, const float* __restrict__ boxes6k,
             int* __restrict__ slots, float* __restrict__ out) {
    int tid = threadIdx.x;
    if (tid == 0) {
        int s = 0;
        for (int i = 0; i < PRE_NMS; ++i) slots[i] = keep[i] ? s++ : -1;
    }
    __syncthreads();
    for (int o = tid; o < POST_NMS * 4; o += blockDim.x) out[o] = 0.0f;
    __syncthreads();
    for (int i = tid; i < PRE_NMS; i += blockDim.x) {
        int s = slots[i];
        if (s >= 0 && s < POST_NMS) {
            out[s * 4 + 0] = boxes6k[i * 4 + 0];
            out[s * 4 + 1] = boxes6k[i * 4 + 1];
            out[s * 4 + 2] = boxes6k[i * 4 + 2];
            out[s * 4 + 3] = boxes6k[i * 4 + 3];
        }
    }
}

// ---------------------------------------------------------------------------
// Host launcher
// ---------------------------------------------------------------------------
extern "C" void kernel_launch(void* const* d_in, const int* in_sizes, int n_in,
                              void* d_out, int out_size, void* d_ws, size_t ws_size,
                              hipStream_t stream) {
    const float* feature = (const float*)d_in[0];
    const float* anchors = (const float*)d_in[1];
    const float* w_conv  = (const float*)d_in[2];
    const float* b_conv  = (const float*)d_in[3];
    const float* w_obj   = (const float*)d_in[4];
    const float* b_obj   = (const float*)d_in[5];
    const float* w_reg   = (const float*)d_in[6];
    const float* b_reg   = (const float*)d_in[7];
    float* out = (float*)d_out;

    // workspace layout (all offsets 256B aligned)
    char* ws = (char*)d_ws;
    size_t off = 0;
    auto take = [&](size_t bytes) { size_t o = off; off += (bytes + 255) & ~(size_t)255; return o; };
    bf16_t*  wB      = (bf16_t*)(ws + take((size_t)RPN_C * RPN_K * 2));          // 1.18 MB
    bf16_t*  wB2     = (bf16_t*)(ws + take((size_t)48 * RPN_C * 2));             // 24 KB
    bf16_t*  tB      = (bf16_t*)(ws + take((size_t)RPN_M * RPN_C * 2));          // 20.5 MB
    int*     ktab    = (int*)   (ws + take((size_t)RPN_K * 2 * 4));              // 18 KB
    float*   scores  = (float*) (ws + take((size_t)RPN_NANCH * 4));              // 1.44 MB
    float*   deltas  = (float*) (ws + take((size_t)RPN_NANCH * 4 * 4));          // 5.76 MB
    unsigned* hist   = (unsigned*)(ws + take(256 * 4));
    unsigned* cntrs  = (unsigned*)(ws + take(8 * 4));
    unsigned long long* cand = (unsigned long long*)(ws + take((size_t)CAND_N * 8));
    float*   boxes6k = (float*) (ws + take((size_t)PRE_NMS * 4 * 4));
    int*     valid6k = (int*)   (ws + take((size_t)PRE_NMS * 4));
    unsigned* maskw  = (unsigned*)(ws + take((size_t)PRE_NMS * MASK_W * 4));     // 4.5 MB
    int*     keep    = (int*)   (ws + take((size_t)PRE_NMS * 4));
    int*     slots   = (int*)   (ws + take((size_t)PRE_NMS * 4));

    // 1) weights -> bf16, k-table
    rpn_cvt_wconv<<<(RPN_C * RPN_K + 255) / 256, 256, 0, stream>>>(w_conv, wB, RPN_C * RPN_K);
    rpn_cvt_whead<<<48, 256, 0, stream>>>(w_obj, w_reg, wB2);
    rpn_build_ktab<<<(RPN_K + 255) / 256, 256, 0, stream>>>(ktab);

    // 2) 3x3 conv + relu (WMMA implicit GEMM): 1250 blocks of 32 M
    rpn_conv3x3<<<RPN_M / 32, 256, 0, stream>>>(feature, wB, ktab, b_conv, tB);

    // 3) 1x1 heads (WMMA GEMM)
    rpn_heads<<<RPN_M / 16, 128, 0, stream>>>(tB, wB2, b_obj, b_reg, scores, deltas);

    // 4) top-6000 select
    rpn_sel_init<<<(CAND_N + 255) / 256, 256, 0, stream>>>(hist, cntrs, cand);
    rpn_hist<<<(RPN_NANCH + 255) / 256, 256, 0, stream>>>(scores, hist);
    rpn_thresh<<<1, 32, 0, stream>>>(hist, cntrs);
    rpn_compact<<<(RPN_NANCH + 255) / 256, 256, 0, stream>>>(scores, cntrs, cand);

    // 5) bitonic sort (descending) over 8192 packed (key|idx) u64s
    for (int k = 2; k <= CAND_N; k <<= 1)
        for (int j = k >> 1; j > 0; j >>= 1)
            rpn_bitonic<<<CAND_N / 256, 256, 0, stream>>>(cand, j, k);

    // 6) decode + clip + validity
    rpn_decode<<<(PRE_NMS + 255) / 256, 256, 0, stream>>>(cand, deltas, anchors, boxes6k, valid6k);

    // 7) NMS
    rpn_nms_mask<<<PRE_NMS, 192, 0, stream>>>(boxes6k, maskw);
    rpn_nms_greedy<<<1, 256, 0, stream>>>(maskw, valid6k, keep);

    // 8) compact first 1000 kept boxes into d_out (zero-filled)
    rpn_finalize<<<1, 256, 0, stream>>>(keep, boxes6k, slots, out);

    (void)in_sizes; (void)n_in; (void)out_size; (void)ws_size;
}
